// BiGNNLayer_23098334118568
// MI455X (gfx1250) — matrix operand order repaired and checked by reading, not compiled, not run on hardware
//
#include <hip/hip_runtime.h>

#define DIM 64

typedef __attribute__((ext_vector_type(16))) _Float16 v16h;
typedef __attribute__((ext_vector_type(8)))  _Float16 v8h;
typedef __attribute__((ext_vector_type(2)))  _Float16 h2;
typedef __attribute__((ext_vector_type(8)))  float    v8f;

// ---------------------------------------------------------------------------
// Phase 0: zero the accumulator workspace x[N,64]
// ---------------------------------------------------------------------------
__global__ __launch_bounds__(256) void zero_ws_kernel(float4* __restrict__ x, int n4) {
    int i = blockIdx.x * 256 + threadIdx.x;
    if (i < n4) x[i] = make_float4(0.f, 0.f, 0.f, 0.f);
}

// ---------------------------------------------------------------------------
// Phase 1: COO SpMM scatter  x[row] += val * features[col]
// One wave32 per edge; each lane owns 2 consecutive dims (64 dims / 32 lanes).
// Non-returning relaxed agent-scope float atomics -> global_atomic_add_f32.
// ---------------------------------------------------------------------------
__global__ __launch_bounds__(256) void spmm_scatter_kernel(
    const int*   __restrict__ rows,
    const int*   __restrict__ cols,
    const float* __restrict__ vals,
    const float* __restrict__ feat,
    float*       __restrict__ x,
    int nE) {
    int e = blockIdx.x * 8 + (threadIdx.x >> 5);   // 8 waves per block
    if (e >= nE) return;
    int lane = threadIdx.x & 31;
    int r = rows[e];
    int c = cols[e];
    float v = vals[e];
    const float2 f2 = *(const float2*)(feat + (long)c * DIM + lane * 2);
    float* dst = x + (long)r * DIM + lane * 2;
    __hip_atomic_fetch_add(dst,     v * f2.x, __ATOMIC_RELAXED, __HIP_MEMORY_SCOPE_AGENT);
    __hip_atomic_fetch_add(dst + 1, v * f2.y, __ATOMIC_RELAXED, __HIP_MEMORY_SCOPE_AGENT);
}

// ---------------------------------------------------------------------------
// Phase 2: out = (f + x) @ W1 + (f * x) @ W2 + (b1 + b2)
// One wave32 per 16-row node tile. f16 WMMA 16x16x32, K=64 -> 2 k-steps.
//
// Weights staged in LDS TRANSPOSED as f16:  sWT[w][n][k] = W[k][n].
// B-fragment layout (ISA 7.12.2): lane L holds column N = L&15, 16
// K values, K = s*32 + 8*(L>=16) + {0..7, 16..23}  ->  two 16-byte runs in the
// transposed layout => two ds_load_b128 per fragment (vs 16 ds_load_u16).
//
// A-fragment layout (16-bit A 16x32): lane L: M = L&15; lanes 0-15 hold
// K = {0..7, 16..23}+s*32, lanes 16-31 hold K = {8..15, 24..31}+s*32.
// C/D layout: VGPR r <-> M = r + 8*(L>=16), N = L&15.
// ---------------------------------------------------------------------------
__global__ __launch_bounds__(256) void fused_wmma_kernel(
    const float* __restrict__ feat,
    const float* __restrict__ x,
    const float* __restrict__ W1,
    const float* __restrict__ b1,
    const float* __restrict__ W2,
    const float* __restrict__ b2,
    float*       __restrict__ out,
    int ntiles) {
    __shared__ _Float16 sWT[2 * DIM * DIM];  // transposed [w][n][k] halfs, 16 KB

    // cooperative transpose-stage: 2048 (n, k-pair) units per matrix,
    // 256 threads x 8 units, packed f16 pair -> one ds_store_b32 each
    const int tid = threadIdx.x;
#pragma unroll
    for (int i = 0; i < 8; ++i) {
        const int u = tid * 8 + i;           // 0..2047
        const int n = u >> 5;                // 0..63
        const int k = (u & 31) * 2;          // even k
        h2 p1 = { (_Float16)W1[k * DIM + n], (_Float16)W1[(k + 1) * DIM + n] };
        h2 p2 = { (_Float16)W2[k * DIM + n], (_Float16)W2[(k + 1) * DIM + n] };
        *(h2*)(sWT + n * DIM + k)             = p1;
        *(h2*)(sWT + DIM * DIM + n * DIM + k) = p2;
    }
    __syncthreads();

    const int wave = tid >> 5;
    const int lane = tid & 31;
    const int tile = blockIdx.x * 8 + wave;
    if (tile >= ntiles) return;              // wave-uniform exit, after barrier

    const int m    = lane & 15;              // A row in tile / B,C,D column N
    const int hi   = lane >> 4;              // lane-group select
    const int row  = tile * 16 + m;
    const int koff = hi * 8;                 // A-matrix K base offset per group

    // Load this lane's 32 f32 values of f and x (4 contiguous chunks of 8)
    float ffl[32], xfl[32];
#pragma unroll
    for (int c = 0; c < 4; ++c) {
        const int kc = c * 16 + koff;        // {0,16,32,48} + {0|8}
        const float4 fa = *(const float4*)(feat + (long)row * DIM + kc);
        const float4 fb = *(const float4*)(feat + (long)row * DIM + kc + 4);
        const float4 xa = *(const float4*)(x    + (long)row * DIM + kc);
        const float4 xb = *(const float4*)(x    + (long)row * DIM + kc + 4);
        ffl[c * 8 + 0] = fa.x; ffl[c * 8 + 1] = fa.y; ffl[c * 8 + 2] = fa.z; ffl[c * 8 + 3] = fa.w;
        ffl[c * 8 + 4] = fb.x; ffl[c * 8 + 5] = fb.y; ffl[c * 8 + 6] = fb.z; ffl[c * 8 + 7] = fb.w;
        xfl[c * 8 + 0] = xa.x; xfl[c * 8 + 1] = xa.y; xfl[c * 8 + 2] = xa.z; xfl[c * 8 + 3] = xa.w;
        xfl[c * 8 + 4] = xb.x; xfl[c * 8 + 5] = xb.y; xfl[c * 8 + 6] = xb.z; xfl[c * 8 + 7] = xb.w;
    }

    // Build A fragments for both k-steps: A1 = f + x, A2 = f * x
    v16h a1[2], a2[2];
#pragma unroll
    for (int s = 0; s < 2; ++s) {
#pragma unroll
        for (int j = 0; j < 8; ++j) {
            const float f0 = ffl[(2 * s) * 8 + j],     x0 = xfl[(2 * s) * 8 + j];
            const float f1 = ffl[(2 * s + 1) * 8 + j], x1 = xfl[(2 * s + 1) * 8 + j];
            a1[s][j]     = (_Float16)(f0 + x0);
            a2[s][j]     = (_Float16)(f0 * x0);
            a1[s][8 + j] = (_Float16)(f1 + x1);
            a2[s][8 + j] = (_Float16)(f1 * x1);
        }
    }

    const int kb_hi = hi * 8;                // B-matrix K base per lane group
#pragma unroll
    for (int nt = 0; nt < 4; ++nt) {
        const float bsum = b1[nt * 16 + m] + b2[nt * 16 + m];
        v8f acc = {bsum, bsum, bsum, bsum, bsum, bsum, bsum, bsum};
        const _Float16* w1p = sWT + (nt * 16 + m) * DIM;             // row n of W1^T
        const _Float16* w2p = sWT + DIM * DIM + (nt * 16 + m) * DIM; // row n of W2^T
#pragma unroll
        for (int s = 0; s < 2; ++s) {
            const int kb = s * 32 + kb_hi;   // two 16B-aligned K runs: kb..kb+7, kb+16..kb+23
            const v8h b1lo = *(const v8h*)(w1p + kb);
            const v8h b1hi = *(const v8h*)(w1p + kb + 16);
            const v8h b2lo = *(const v8h*)(w2p + kb);
            const v8h b2hi = *(const v8h*)(w2p + kb + 16);
            const v16h bw1 = __builtin_shufflevector(b1lo, b1hi,
                0, 1, 2, 3, 4, 5, 6, 7, 8, 9, 10, 11, 12, 13, 14, 15);
            const v16h bw2 = __builtin_shufflevector(b2lo, b2hi,
                0, 1, 2, 3, 4, 5, 6, 7, 8, 9, 10, 11, 12, 13, 14, 15);
            acc = __builtin_amdgcn_wmma_f32_16x16x32_f16(
                false, a1[s], false, bw1, (short)0, acc, false, false);
            acc = __builtin_amdgcn_wmma_f32_16x16x32_f16(
                false, a2[s], false, bw2, (short)0, acc, false, false);
        }
#pragma unroll
        for (int r = 0; r < 8; ++r) {
            const int mm = r + hi * 8;
            out[(long)(tile * 16 + mm) * DIM + nt * 16 + m] = acc[r];
        }
    }
}

// ---------------------------------------------------------------------------
extern "C" void kernel_launch(void* const* d_in, const int* in_sizes, int n_in,
                              void* d_out, int out_size, void* d_ws, size_t ws_size,
                              hipStream_t stream) {
    const int*   rows = (const int*)  d_in[0];
    const int*   cols = (const int*)  d_in[1];
    const float* vals = (const float*)d_in[2];
    const float* feat = (const float*)d_in[3];
    const float* W1   = (const float*)d_in[4];
    const float* b1   = (const float*)d_in[5];
    const float* W2   = (const float*)d_in[6];
    const float* b2   = (const float*)d_in[7];
    float* out = (float*)d_out;

    const int E = in_sizes[0];
    const int N = in_sizes[3] / DIM;
    float* x = (float*)d_ws;                 // N*DIM*4 = 25.6 MB scratch

    const int n4 = (N * DIM) / 4;
    zero_ws_kernel<<<(n4 + 255) / 256, 256, 0, stream>>>((float4*)x, n4);

    spmm_scatter_kernel<<<(E + 7) / 8, 256, 0, stream>>>(rows, cols, vals, feat, x, E);

    const int ntiles = (N + 15) / 16;        // N=100000 -> 6250 exact
    fused_wmma_kernel<<<(ntiles + 7) / 8, 256, 0, stream>>>(
        feat, x, W1, b1, W2, b2, out, ntiles);
}
